// NCSSMVision_6090263626442
// MI455X (gfx1250) — matrix operand order repaired and checked by reading, not compiled, not run on hardware
//
#include <hip/hip_runtime.h>
#include <hip/hip_bf16.h>
#include <math.h>

typedef __attribute__((ext_vector_type(16))) _Float16 v16h;
typedef __attribute__((ext_vector_type(8)))  float    v8f;
typedef __attribute__((ext_vector_type(4)))  unsigned int v4u;
typedef __attribute__((ext_vector_type(8)))  int      v8i;
typedef __attribute__((ext_vector_type(4)))  int      v4i;

#define DI    256      // D_INNER
#define NS    16       // D_STATE
#define LSEQ  4096
#define BATCH 8
#define KCOLS 33       // 2N+1
#define KPAD  48       // padded to 3 x 16 WMMA column tiles
#define MROWS (BATCH*LSEQ)
#define CONST_FLOATS 1024

#if defined(__has_builtin)
# if __has_builtin(__builtin_amdgcn_tensor_load_to_lds) && __has_builtin(__builtin_amdgcn_s_wait_tensorcnt)
#  define USE_TDM 1
# endif
#endif
#ifndef USE_TDM
# define USE_TDM 0
#endif

// -------------------------------------------------------------------
// Tensor Data Mover helpers (D# per CDNA5 ISA ch.8).
// group0: [1:0]=count=1, [63:32]=lds_addr, [120:64]=global_addr, [127:126]=type(2)
// group1: [17:16]=data_size(2 => 4B), [79:48]=tensor_dim0, [111:80]=tensor_dim1,
//         [127:112]=tile_dim0, [143:128]=tile_dim1, [207:160]=tensor_dim0_stride
// -------------------------------------------------------------------
#if USE_TDM
static __device__ __forceinline__ unsigned lds_off(const void* p) {
    // Low 32 bits of a flat pointer into LDS are the LDS byte offset.
    return (unsigned)(unsigned long long)p;
}

static __device__ __forceinline__ void tdm_issue(v4u g0, v8i g1) {
    v4i z = {0, 0, 0, 0};
#if __clang_major__ >= 23
    v8i z8 = {0, 0, 0, 0, 0, 0, 0, 0};
    __builtin_amdgcn_tensor_load_to_lds(g0, g1, z, z, z8, 0);
#else
    __builtin_amdgcn_tensor_load_to_lds(g0, g1, z, z, 0);
#endif
}

// 2D tile of float32: 'w' elems/row, 'h' rows, global row stride 'srcStride'
// elems; packed contiguously into LDS at byte offset 'ldsOff'.
static __device__ __forceinline__ void tdm_load_f32_2d(
    unsigned ldsOff, const float* gptr, unsigned w, unsigned h, unsigned srcStride) {
    unsigned long long ga = (unsigned long long)gptr;
    v4u g0;
    g0[0] = 1u;                                               // count=1
    g0[1] = ldsOff;
    g0[2] = (unsigned)(ga & 0xFFFFFFFFull);
    g0[3] = (unsigned)((ga >> 32) & 0x01FFFFFFull) | (2u << 30); // type=2
    v8i g1;
    g1[0] = (int)(2u << 16);                                  // data_size = 4B
    g1[1] = (int)((w & 0xFFFFu) << 16);                       // tensor_dim0 lo16
    g1[2] = (int)(((w >> 16) & 0xFFFFu) | ((h & 0xFFFFu) << 16)); // dim0 hi | dim1 lo
    g1[3] = (int)(((h >> 16) & 0xFFFFu) | ((w & 0xFFFFu) << 16)); // dim1 hi | tile_dim0
    g1[4] = (int)(h & 0xFFFFu);                               // tile_dim1 (0 => 1D)
    g1[5] = (int)srcStride;                                   // dim0_stride lo32
    g1[6] = 0;
    g1[7] = 0;
    tdm_issue(g0, g1);
}

// 1D contiguous copy of n float32 elements.
static __device__ __forceinline__ void tdm_load_f32_1d(
    unsigned ldsOff, const float* gptr, unsigned n) {
    unsigned long long ga = (unsigned long long)gptr;
    v4u g0;
    g0[0] = 1u;
    g0[1] = ldsOff;
    g0[2] = (unsigned)(ga & 0xFFFFFFFFull);
    g0[3] = (unsigned)((ga >> 32) & 0x01FFFFFFull) | (2u << 30);
    v8i g1;
    g1[0] = (int)(2u << 16);                                  // data_size = 4B
    g1[1] = (int)((n & 0xFFFFu) << 16);                       // tensor_dim0 lo16
    g1[2] = (int)(((n >> 16) & 0xFFFFu) | (1u << 16));        // dim0 hi | tensor_dim1=1
    g1[3] = (int)((n & 0xFFFFu) << 16);                       // tile_dim0 = n
    g1[4] = 0;                                                // tile_dim1=0 -> 1D
    g1[5] = (int)n;                                           // dim0_stride
    g1[6] = 0;
    g1[7] = 0;
    tdm_issue(g0, g1);
}
#endif // USE_TDM

// -------------------------------------------------------------------
// Kernel 1: per-batch vis-derived constants (everything constant in L)
// -------------------------------------------------------------------
__global__ __launch_bounds__(128) void vis_consts_kernel(
    const float* __restrict__ vis, const float* __restrict__ A_log,
    const float* __restrict__ B_base, const float* __restrict__ C_base,
    const float* __restrict__ half_sat,
    const float* __restrict__ dfmin, const float* __restrict__ dfmax,
    const float* __restrict__ bgf,   const float* __restrict__ alpha,
    const float* __restrict__ emin,  const float* __restrict__ emax,
    const float* __restrict__ dt_base,
    const float* __restrict__ sbias_dt, const float* __restrict__ sbias_BC,
    const float* __restrict__ sscale,   const float* __restrict__ Bsu,
    float* __restrict__ consts)
{
    __shared__ float sub[BATCH][NS];
    const int tid = threadIdx.x;
    const int b = tid >> 4, n = tid & 15;

    const float hs = fabsf(*half_sat);
    float acc = 0.f;
    #pragma unroll
    for (int p = 0; p < 12; ++p) {
        float v = vis[b * 192 + n * 12 + p];
        v = fminf(fmaxf(v, 0.f), 1.f);
        acc += v / (v + hs);
    }
    const float vs = acc * (1.f / 12.f);
    sub[b][n] = vs;
    __syncthreads();

    float m = 0.f;
    #pragma unroll
    for (int i = 0; i < NS; ++i) m += sub[b][i];
    m *= (1.f / NS);
    float var = 0.f;
    #pragma unroll
    for (int i = 0; i < NS; ++i) { float d = sub[b][i] - m; var += d * d; }
    var *= (1.f / NS);

    const float cv   = fminf(fmaxf(sqrtf(var + 1e-6f) / (fabsf(m) + 1e-4f), 0.f), 1.f);
    const float unif = 1.f - cv;

    const float sigBC = 1.f / (1.f + __expf(-(sscale[n] * vs + sbias_BC[n])));
    const float Bmod  = B_base[n] * (1.f + fminf(fmaxf(Bsu[n], -5.f), 5.f) * unif);
    const float al    = *alpha;
    const float gate  = vs * (1.f - *bgf) + *bgf;
    const float gmul  = 1.f - al + al * gate;

    consts[32  + b * NS + n] = sigBC * gmul;
    consts[160 + b * NS + n] = (1.f - sigBC) * Bmod * gmul;
    consts[288 + b * NS + n] = sigBC;
    consts[416 + b * NS + n] = (1.f - sigBC) * C_base[n];

    if (n == 0) {
        const float sig = 1.f / (1.f + __expf(-(5.f * m + *sbias_dt)));
        consts[b]      = sig;
        consts[8 + b]  = (1.f - sig) * dt_base[0];
        consts[16 + b] = *dfmin + (*dfmax - *dfmin) * m;   // adaptive_floor
        consts[24 + b] = *emax  - (*emax - *emin) * m;     // adaptive_eps
    }
    if (b == 0) consts[544 + n] = -__expf(A_log[n]);
}

// -------------------------------------------------------------------
// Kernel 2: xp GEMM via WMMA (f16 in, f32 accum) + fused param epilogue
// -------------------------------------------------------------------
__global__ __launch_bounds__(128) void gemm_params_kernel(
    const float* __restrict__ x, const float* __restrict__ W,
    const float* __restrict__ dt_w, const float* __restrict__ dt_b,
    const float* __restrict__ consts, float* __restrict__ params)
{
    __shared__ __align__(32) _Float16 Wl[KPAD * DI];   // 24KB
    __shared__ float tile[4][16][KPAD];                // 12KB

    const int tid = threadIdx.x;

    for (int i = tid; i < KCOLS * DI; i += 128) Wl[i] = (_Float16)W[i];
    for (int i = KCOLS * DI + tid; i < KPAD * DI; i += 128) Wl[i] = (_Float16)0.f;
    __syncthreads();

    const int wave = tid >> 5, lane = tid & 31;
    const int lrow = lane & 15;
    const int hi   = lane >> 4;
    const int tileIdx = blockIdx.x * 4 + wave;
    const int row0 = tileIdx * 16;

    v8f c0 = {}, c1 = {}, c2 = {};
    const float* xrow = x + (size_t)(row0 + lrow) * DI;

    for (int kk = 0; kk < DI; kk += 32) {
        const int base = kk + hi * 16;
        const float4* p0 = (const float4*)(xrow + base);
        const float4* p1 = (const float4*)(xrow + base + 8);
        float4 f0 = p0[0], f1 = p0[1], f2 = p1[0], f3 = p1[1];
        v16h a;
        a[0] = (_Float16)f0.x;  a[1] = (_Float16)f0.y;  a[2] = (_Float16)f0.z;  a[3] = (_Float16)f0.w;
        a[4] = (_Float16)f1.x;  a[5] = (_Float16)f1.y;  a[6] = (_Float16)f1.z;  a[7] = (_Float16)f1.w;
        a[8] = (_Float16)f2.x;  a[9] = (_Float16)f2.y;  a[10] = (_Float16)f2.z; a[11] = (_Float16)f2.w;
        a[12] = (_Float16)f3.x; a[13] = (_Float16)f3.y; a[14] = (_Float16)f3.z; a[15] = (_Float16)f3.w;

        const v16h* b0 = (const v16h*)&Wl[(0 * 16 + lrow) * DI + kk + hi * 16];
        const v16h* b1 = (const v16h*)&Wl[(1 * 16 + lrow) * DI + kk + hi * 16];
        const v16h* b2 = (const v16h*)&Wl[(2 * 16 + lrow) * DI + kk + hi * 16];
        c0 = __builtin_amdgcn_wmma_f32_16x16x32_f16(false, a, false, *b0, (short)0, c0, false, false);
        c1 = __builtin_amdgcn_wmma_f32_16x16x32_f16(false, a, false, *b1, (short)0, c1, false, false);
        c2 = __builtin_amdgcn_wmma_f32_16x16x32_f16(false, a, false, *b2, (short)0, c2, false, false);
    }

    #pragma unroll
    for (int v = 0; v < 8; ++v) {
        tile[wave][v + 8 * hi][0 * 16 + lrow] = c0[v];
        tile[wave][v + 8 * hi][1 * 16 + lrow] = c1[v];
        tile[wave][v + 8 * hi][2 * 16 + lrow] = c2[v];
    }
    __syncthreads();

    if (lane < 16) {
        const int row = row0 + lrow;          // = b*LSEQ + l
        const int b = row >> 12;
        const float* tp = tile[wave][lrow];

        const float sig = consts[b], q0 = consts[8 + b];
        const float fl  = consts[16 + b], ep = consts[24 + b];
        const float dtw = *dt_w, dtb = *dt_b;

        const float dtsel = tp[0];
        const float z  = (sig * dtsel + q0) * dtw + dtb;
        const float sp = (z > 20.f) ? z : log1pf(__expf(z));
        const float delta = fminf(sp, 1.0f) + fl;

        float* outp = params + (size_t)row * KPAD;
        #pragma unroll
        for (int n = 0; n < NS; ++n) {
            const float dA = __expf(consts[544 + n] * delta);
            const float Bp = consts[32  + b * NS + n] * tp[1 + n]  + consts[160 + b * NS + n];
            const float Cp = consts[288 + b * NS + n] * tp[17 + n] + consts[416 + b * NS + n];
            outp[n]      = dA;
            outp[16 + n] = delta * Bp + ep;
            outp[32 + n] = Cp;
        }
    }
}

// -------------------------------------------------------------------
// Kernel 3: sequential selective scan. One wave per block, one d per
// lane, h[16] in VGPRs. Double-buffered TDM (tensor_load_to_lds) stages
// next chunk's params (1D, 6KB) and x tile (2D 32x32, stride D) while
// the wave computes the current chunk; drain via s_wait_tensorcnt.
// -------------------------------------------------------------------
#define CH 32
__global__ __launch_bounds__(32) void scan_kernel(
    const float* __restrict__ x, const float* __restrict__ params,
    const float* __restrict__ Dskip, float* __restrict__ out)
{
    __shared__ __align__(16) float ps[2][CH][KPAD];   // 2 x 6KB
    __shared__ __align__(16) float xs[2][CH][32];     // 2 x 4KB

    const int tid   = threadIdx.x;
    const int b     = blockIdx.x >> 3;
    const int dbase = (blockIdx.x & 7) << 5;
    const int d     = dbase + tid;

    const float* xbB = x      + ((size_t)b * LSEQ) * DI + dbase;   // block base
    const float* pb  = params + ((size_t)b * LSEQ) * KPAD;
    float*       ob  = out    + ((size_t)b * LSEQ) * DI + d;
    const float dsk  = Dskip[d];

    float h[NS];
    #pragma unroll
    for (int n = 0; n < NS; ++n) h[n] = 0.f;

#if USE_TDM
    // Prologue: DMA chunk 0 into buffer 0.
    tdm_load_f32_1d(lds_off(&ps[0][0][0]), pb, CH * KPAD);
    tdm_load_f32_2d(lds_off(&xs[0][0][0]), xbB, 32, CH, DI);
    __builtin_amdgcn_s_wait_tensorcnt(0);
    __asm__ volatile("" ::: "memory");
#endif

    for (int l0 = 0; l0 < LSEQ; l0 += CH) {
        const int cur = (l0 / CH) & 1;
        const int nxt = cur ^ 1;

#if USE_TDM
        // Kick off DMA for the next chunk while we compute this one.
        if (l0 + CH < LSEQ) {
            tdm_load_f32_1d(lds_off(&ps[nxt][0][0]), pb + (size_t)(l0 + CH) * KPAD,
                            CH * KPAD);
            tdm_load_f32_2d(lds_off(&xs[nxt][0][0]), xbB + (size_t)(l0 + CH) * DI,
                            32, CH, DI);
        }
#else
        __syncthreads();
        {
            const float4* src = (const float4*)(pb + (size_t)l0 * KPAD);
            float4* dst = (float4*)&ps[cur][0][0];
            #pragma unroll 4
            for (int i = tid; i < CH * KPAD / 4; i += 32) dst[i] = src[i];
        }
        #pragma unroll 4
        for (int i = 0; i < CH; ++i) xs[cur][i][tid] = xbB[(size_t)(l0 + i) * DI + tid];
        if (l0 + CH < LSEQ) {
            __builtin_prefetch(&xbB[(size_t)(l0 + CH) * DI + tid], 0, 0);
            __builtin_prefetch(&pb[(size_t)(l0 + CH) * KPAD + tid * 4], 0, 0);
        }
        __syncthreads();
#endif

        for (int i = 0; i < CH; ++i) {
            const float xv = xs[cur][i][tid];
            float4 q[12];
            const float4* pv = (const float4*)&ps[cur][i][0];
            #pragma unroll
            for (int j = 0; j < 12; ++j) q[j] = pv[j];
            const float* dA = (const float*)&q[0];   // [0..15]
            const float* w  = (const float*)&q[4];   // [16..31]
            const float* C  = (const float*)&q[8];   // [32..47]

            float y = 0.f;
            #pragma unroll
            for (int n = 0; n < NS; ++n) {
                float hn = dA[n] * h[n] + w[n] * xv;
                hn = fminf(fmaxf(hn, -10000.f), 10000.f);
                h[n] = hn;
                y += hn * C[n];
            }
            y += dsk * xv;
            if (!__builtin_isfinite(y))
                y = __builtin_isnan(y) ? 0.f : (y > 0.f ? 10000.f : -10000.f);
            ob[(size_t)(l0 + i) * DI] = y;
        }

#if USE_TDM
        // Drain next-chunk DMA before swapping buffers.
        __builtin_amdgcn_s_wait_tensorcnt(0);
        __asm__ volatile("" ::: "memory");
#endif
    }
}

// -------------------------------------------------------------------
extern "C" void kernel_launch(void* const* d_in, const int* in_sizes, int n_in,
                              void* d_out, int out_size, void* d_ws, size_t ws_size,
                              hipStream_t stream) {
    const float* x        = (const float*)d_in[0];
    const float* vis      = (const float*)d_in[1];
    const float* W        = (const float*)d_in[2];
    const float* dt_w     = (const float*)d_in[3];
    const float* dt_b     = (const float*)d_in[4];
    const float* A_log    = (const float*)d_in[5];
    const float* D_skip   = (const float*)d_in[6];
    const float* half_sat = (const float*)d_in[7];
    const float* dfmin    = (const float*)d_in[8];
    const float* dfmax    = (const float*)d_in[9];
    const float* bgf      = (const float*)d_in[10];
    const float* alpha    = (const float*)d_in[11];
    const float* emin     = (const float*)d_in[12];
    const float* emax     = (const float*)d_in[13];
    const float* dt_base  = (const float*)d_in[14];
    const float* B_base   = (const float*)d_in[15];
    const float* C_base   = (const float*)d_in[16];
    const float* sbias_dt = (const float*)d_in[17];
    const float* sbias_BC = (const float*)d_in[18];
    const float* sscale   = (const float*)d_in[19];
    const float* Bsu      = (const float*)d_in[20];

    float* out    = (float*)d_out;
    float* consts = (float*)d_ws;
    float* params = consts + CONST_FLOATS;

    vis_consts_kernel<<<1, 128, 0, stream>>>(
        vis, A_log, B_base, C_base, half_sat, dfmin, dfmax, bgf, alpha,
        emin, emax, dt_base, sbias_dt, sbias_BC, sscale, Bsu, consts);

    gemm_params_kernel<<<MROWS / 16 / 4, 128, 0, stream>>>(
        x, W, dt_w, dt_b, consts, params);

    scan_kernel<<<BATCH * (DI / 32), 32, 0, stream>>>(x, params, D_skip, out);
}
